// VoidLinear_83004537962674
// MI455X (gfx1250) — matrix operand order repaired and checked by reading, not compiled, not run on hardware
//
#include <hip/hip_runtime.h>

// ---------------------------------------------------------------------------
// VoidLinear for MI455X (gfx1250, wave32, WMMA + async global->LDS DMA)
//
// Pipeline (all on `stream`):
//   1) f32->bf16 convert: input (8192x4096), lowrank_first (128x4096),
//      lowrank_second (8192x128)
//   2) dequant: codebook-gather + per-8x8 affine -> w_bf16 [OUT_F][IN_F]
//   3) hid GEMM (WMMA bf16): hid = act @ lf^T  -> bf16 [8192][128]
//   4) fused main GEMM (WMMA bf16, f32 acc, LDS double-buffered via
//      GLOBAL_LOAD_ASYNC_TO_LDS_B128 / s_wait_asynccnt):
//        baseline = act @ w^T
//        epilogue per 16x16 tile: mult = hid@ls[0:4096]^T + scale
//                                 add  = hid@ls[4096:]^T + bias
//        out = add + mult * baseline     (fp32)
// ---------------------------------------------------------------------------

typedef __attribute__((ext_vector_type(16))) __bf16 v16bf;
typedef __attribute__((ext_vector_type(8)))  __bf16 v8bf;
typedef __attribute__((ext_vector_type(8)))  float  v8f;

#define IN_F   4096
#define OUT_F  4096
#define TOKENS 8192
#define LRK    128
#define BS     64
#define CB     256
#define BK     64          // K elements staged per LDS stage

// ---------------------------------------------------------------------------
// Global-memory fragment loaders (used by hid GEMM + low-rank epilogue).
// A-fragment (16x32 bf16): lane l holds row M = l%16, K-chunks
//   [k0 + 8*(l/16), +8) and [k0 + 16 + 8*(l/16), +8)            (ISA 7.12.2)
// B-fragment (32x16 bf16): lane l holds column N = l%16, K-chunks
//   [k0 + 8*(l/16), +8) and [k0 + 16 + 8*(l/16), +8) with [N][K] row-major weights.
// ---------------------------------------------------------------------------
__device__ __forceinline__ v16bf load_a_frag(const __bf16* __restrict__ base,
                                             int stride, int row0, int k0,
                                             int laneM, int laneH) {
    const __bf16* p = base + (size_t)(row0 + laneM) * stride + k0 + laneH * 8;
    v8bf lo = *(const v8bf*)(p);
    v8bf hi = *(const v8bf*)(p + 16);
    return __builtin_shufflevector(lo, hi, 0, 1, 2, 3, 4, 5, 6, 7,
                                          8, 9, 10, 11, 12, 13, 14, 15);
}

__device__ __forceinline__ v16bf load_b_frag(const __bf16* __restrict__ base,
                                             int stride, int col0, int k0,
                                             int laneM, int laneH) {
    const __bf16* p = base + (size_t)(col0 + laneM) * stride + k0 + laneH * 8;
    v8bf lo = *(const v8bf*)(p);
    v8bf hi = *(const v8bf*)(p + 16);
    return __builtin_shufflevector(lo, hi, 0, 1, 2, 3, 4, 5, 6, 7,
                                          8, 9, 10, 11, 12, 13, 14, 15);
}

__device__ __forceinline__ v8f wmma_bf16(v16bf a, v16bf b, v8f c) {
    return __builtin_amdgcn_wmma_f32_16x16x32_bf16(
        /*neg_a=*/false, a, /*neg_b=*/false, b,
        /*c_mod=*/(short)0, c, /*reuse_a=*/false, /*reuse_b=*/false);
}

// ---------------------------------------------------------------------------
// CDNA5 async global->LDS copy (16B per lane), tracked by ASYNCcnt.
// VDST operand = per-lane LDS byte address (ISA §10 / §15.18, opcode 98).
// ---------------------------------------------------------------------------
__device__ __forceinline__ unsigned int lds_addr_of(const void* p) {
    return (unsigned int)(unsigned long long)
        (__attribute__((address_space(3))) const char*)p;
}

__device__ __forceinline__ void async_copy16(unsigned int lds_byte_addr,
                                             const __bf16* gptr) {
    asm volatile("global_load_async_to_lds_b128 %0, %1, off"
                 :: "v"(lds_byte_addr), "v"(gptr)
                 : "memory");
}

__device__ __forceinline__ void wait_async0() {
    asm volatile("s_wait_asynccnt 0x0" ::: "memory");
}

// ---------------------------------------------------------------------------
// LDS tile layout: "fragment order". For a 128 x BK bf16 tile, the 32-byte
// fragment of (16-row tile t, 32-K substep s, lane l) lives at byte offset
//   ((t*(BK/32) + s)*32 + l)*32
// so each wave's fragment read is one lane-linear 32B ds_load: conflict-free,
// no padding, v16bf-aligned. The global->LDS swizzle is applied on the copy.
//   A chunk (row, kc):  lane = row%16 + 16*(kc&1),     slot = (kc>>1)&1
//   B chunk (row, kc):  lane = row%16 + 16*(kc&1),     slot = (kc>>1)&1
// (kc = 8-element K-chunk index within the stage, 0..BK/8-1)
// ---------------------------------------------------------------------------
__device__ __forceinline__ int frag_off(int tile, int s, int lane) {
    return (((tile * (BK / 32) + s) * 32) + lane) * 32;   // bytes
}

// ---------------------------------------------------------------------------
// Generic f32 -> bf16 convert, 8 elems/thread. n must be a multiple of 2048.
// ---------------------------------------------------------------------------
__global__ void k_f32_to_bf16(const float* __restrict__ src,
                              __bf16* __restrict__ dst, int n) {
    int i = (blockIdx.x * blockDim.x + threadIdx.x) * 8;
    if (i + 8 > n) return;
    float4 a = *(const float4*)(src + i);
    float4 b = *(const float4*)(src + i + 4);
    v8bf o;
    o[0] = (__bf16)a.x; o[1] = (__bf16)a.y; o[2] = (__bf16)a.z; o[3] = (__bf16)a.w;
    o[4] = (__bf16)b.x; o[5] = (__bf16)b.y; o[6] = (__bf16)b.z; o[7] = (__bf16)b.w;
    *(v8bf*)(dst + i) = o;
}

// ---------------------------------------------------------------------------
// Dequantize: rm[o][i] = grid_bias + codebooks[blk][zm[...]] * grid_scale,
// stored bf16 row-major [OUT_F][IN_F]. One thread = 8 consecutive i.
// ---------------------------------------------------------------------------
__global__ void k_dequant(const int* __restrict__ zm,
                          const float* __restrict__ cb,
                          const float* __restrict__ gscale,
                          const float* __restrict__ gbias,
                          __bf16* __restrict__ w) {
    int tid = blockIdx.x * blockDim.x + threadIdx.x;   // OUT_F * IN_F / 8
    int o  = tid >> 9;              // / (IN_F/8)
    int i0 = (tid & 511) << 3;      // i base, multiple of 8
    int ob = o >> 6, r = o & 63;
    int ib = i0 >> 6, c0 = i0 & 63;
    int blk = ob * (IN_F / BS) + ib;
    const int*   zp  = zm + ((size_t)blk * BS + r) * BS + c0;
    const float* cbr = cb + (size_t)blk * CB;
    int g = (o >> 3) * (IN_F / 8) + (i0 >> 3);
    float s = gscale[g];
    float b = gbias[g];
    v8bf ov;
#pragma unroll
    for (int t = 0; t < 8; ++t) {
        float v = b + cbr[zp[t]] * s;
        ov[t] = (__bf16)v;
    }
    *(v8bf*)(w + (size_t)o * IN_F + i0) = ov;
}

// ---------------------------------------------------------------------------
// hid = act(8192x4096) @ lf^T(4096x128) in bf16 WMMA, output bf16 [8192][128].
// One 16x16 tile per wave; 8 waves per block. Everything is L2-resident.
// ---------------------------------------------------------------------------
__global__ void k_hid_gemm(const __bf16* __restrict__ act,
                           const __bf16* __restrict__ lf,
                           __bf16* __restrict__ hid) {
    int lane  = threadIdx.x & 31;
    int wid   = threadIdx.x >> 5;
    int laneM = lane & 15, laneH = lane >> 4;
    int tile = blockIdx.x * 8 + wid;        // (TOKENS/16) * (LRK/16) tiles
    int mt = (tile >> 3) << 4;              // token tile base
    int nt = (tile & 7) << 4;               // low-rank tile base
    v8f acc = {};
#pragma unroll 4
    for (int k0 = 0; k0 < IN_F; k0 += 32) {
        v16bf a = load_a_frag(act, IN_F, mt, k0, laneM, laneH);
        v16bf b = load_b_frag(lf, IN_F, nt, k0, laneM, laneH);
        acc = wmma_bf16(a, b, acc);
    }
    int mrow = mt + laneH * 8;
#pragma unroll
    for (int rr = 0; rr < 8; ++rr)
        hid[(size_t)(mrow + rr) * LRK + nt + laneM] = (__bf16)acc[rr];
}

// ---------------------------------------------------------------------------
// Fused main GEMM, LDS double-buffered.
// Block: 256 threads = 8 waves; block tile 128(M) x 128(N); BK=64 per stage.
// Wave grid 4(M) x 2(N): each wave computes a 32x64 region = 2x4 WMMA tiles.
// Per stage: 8 async 16B copies/thread for the NEXT stage overlap 16 WMMAs
// on the current stage; one s_wait_asynccnt + one barrier per stage.
// LDS: 2 * (16KB A + 16KB B) = 64KB.
// ---------------------------------------------------------------------------
__global__ void __launch_bounds__(256, 2)
k_fused_gemm(const __bf16* __restrict__ act,
             const __bf16* __restrict__ w,
             const __bf16* __restrict__ hid,
             const __bf16* __restrict__ ls,
             const float* __restrict__ scale,
             const float* __restrict__ bias,
             float* __restrict__ out) {
    __shared__ __bf16 As[2][128 * BK];   // 32 KB
    __shared__ __bf16 Bs[2][128 * BK];   // 32 KB

    int tid   = threadIdx.x;
    int lane  = tid & 31;
    int wid   = tid >> 5;
    int laneM = lane & 15, laneH = lane >> 4;
    int waveM = wid & 3;           // 4 waves along M
    int waveN = wid >> 2;          // 2 waves along N
    int mBlock = blockIdx.y * 128;
    int nBlock = blockIdx.x * 128;
    int mBase = mBlock + waveM * 32;
    int nBase = nBlock + waveN * 64;

    // Per-thread cooperative-copy geometry: 4 chunks each for A and B.
    // chunk c (0..1023): row = c>>3, kc = c&7.
    int aLdsOff[4], bLdsOff[4];
    const __bf16* aGlb[4];
    const __bf16* bGlb[4];
#pragma unroll
    for (int p = 0; p < 4; ++p) {
        int c   = tid + p * 256;
        int row = c >> 3;
        int kc  = c & 7;
        int t   = row >> 4;
        int aLane = (row & 15) + 16 * (kc & 1);
        int aSlot = (kc >> 1) & 1;
        int bLane = (row & 15) + 16 * (kc & 1);
        int bSlot = (kc >> 1) & 1;
        aLdsOff[p] = frag_off(t, kc >> 2, aLane) + aSlot * 16;
        bLdsOff[p] = frag_off(t, kc >> 2, bLane) + bSlot * 16;
        aGlb[p] = act + (size_t)(mBlock + row) * IN_F + kc * 8;
        bGlb[p] = w   + (size_t)(nBlock + row) * IN_F + kc * 8;
    }

    v8f acc[2][4];
#pragma unroll
    for (int i = 0; i < 2; ++i)
#pragma unroll
        for (int j = 0; j < 4; ++j) acc[i][j] = (v8f){};

    // ---- prologue: stage 0 ------------------------------------------------
    unsigned int aBase0 = lds_addr_of(&As[0][0]);
    unsigned int bBase0 = lds_addr_of(&Bs[0][0]);
    unsigned int aBase1 = lds_addr_of(&As[1][0]);
    unsigned int bBase1 = lds_addr_of(&Bs[1][0]);
#pragma unroll
    for (int p = 0; p < 4; ++p) {
        async_copy16(aBase0 + aLdsOff[p], aGlb[p]);
        async_copy16(bBase0 + bLdsOff[p], bGlb[p]);
    }
    wait_async0();
    __syncthreads();

    const int nStages = IN_F / BK;     // 64
    for (int st = 0; st < nStages; ++st) {
        int buf = st & 1;
        // issue async copies for the next stage into the other buffer
        if (st + 1 < nStages) {
            unsigned int aB = buf ? aBase0 : aBase1;
            unsigned int bB = buf ? bBase0 : bBase1;
            int kOff = (st + 1) * BK;
#pragma unroll
            for (int p = 0; p < 4; ++p) {
                async_copy16(aB + aLdsOff[p], aGlb[p] + kOff);
                async_copy16(bB + bLdsOff[p], bGlb[p] + kOff);
            }
        }
        // compute on current buffer: 2 K-substeps x 8 WMMAs
        const char* Ab = (const char*)As[buf];
        const char* Bb = (const char*)Bs[buf];
#pragma unroll
        for (int s = 0; s < 2; ++s) {
            v16bf a0 = *(const v16bf*)(Ab + frag_off(2 * waveM + 0, s, lane));
            v16bf a1 = *(const v16bf*)(Ab + frag_off(2 * waveM + 1, s, lane));
            v16bf b0 = *(const v16bf*)(Bb + frag_off(4 * waveN + 0, s, lane));
            v16bf b1 = *(const v16bf*)(Bb + frag_off(4 * waveN + 1, s, lane));
            v16bf b2 = *(const v16bf*)(Bb + frag_off(4 * waveN + 2, s, lane));
            v16bf b3 = *(const v16bf*)(Bb + frag_off(4 * waveN + 3, s, lane));
            acc[0][0] = wmma_bf16(a0, b0, acc[0][0]);
            acc[0][1] = wmma_bf16(a0, b1, acc[0][1]);
            acc[0][2] = wmma_bf16(a0, b2, acc[0][2]);
            acc[0][3] = wmma_bf16(a0, b3, acc[0][3]);
            acc[1][0] = wmma_bf16(a1, b0, acc[1][0]);
            acc[1][1] = wmma_bf16(a1, b1, acc[1][1]);
            acc[1][2] = wmma_bf16(a1, b2, acc[1][2]);
            acc[1][3] = wmma_bf16(a1, b3, acc[1][3]);
        }
        if (st + 1 < nStages) wait_async0();
        __syncthreads();
    }

    // ---- fused low-rank epilogue (K=128, L2-resident, direct global) -----
#pragma unroll
    for (int i = 0; i < 2; ++i) {
        int mt = mBase + i * 16;
#pragma unroll
        for (int j = 0; j < 4; ++j) {
            int nt = nBase + j * 16;
            v8f mAcc = {};
            v8f aAcc = {};
#pragma unroll
            for (int kk = 0; kk < LRK; kk += 32) {
                v16bf ah = load_a_frag(hid, LRK, mt, kk, laneM, laneH);
                v16bf bm = load_b_frag(ls, LRK, nt,         kk, laneM, laneH);
                v16bf ba = load_b_frag(ls, LRK, OUT_F + nt, kk, laneM, laneH);
                mAcc = wmma_bf16(ah, bm, mAcc);
                aAcc = wmma_bf16(ah, ba, aAcc);
            }
            int n    = nt + laneM;
            float s  = scale[n];
            float bi = bias[n];
            int mrow = mt + laneH * 8;
#pragma unroll
            for (int rr = 0; rr < 8; ++rr) {
                float mult = mAcc[rr] + s;
                float addv = aAcc[rr] + bi;
                out[(size_t)(mrow + rr) * OUT_F + n] = addv + mult * acc[i][j][rr];
            }
        }
    }
}

// ---------------------------------------------------------------------------
// Host launcher.
// Inputs (dict order): 0 input, 1 zm, 2 codebooks, 3 lowrank_first,
//                      4 lowrank_second, 5 grid_scale, 6 grid_bias,
//                      7 scale, 8 bias
// Workspace layout (bytes):
//   act_bf16 [8192x4096]  @ 0          (64 MB)
//   w_bf16   [4096x4096]  @ 67108864   (32 MB)
//   lf_bf16  [128x4096]   @ 100663296  ( 1 MB)
//   ls_bf16  [8192x128]   @ 101711872  ( 2 MB)
//   hid_bf16 [8192x128]   @ 103809024  ( 2 MB)
// ---------------------------------------------------------------------------
extern "C" void kernel_launch(void* const* d_in, const int* in_sizes, int n_in,
                              void* d_out, int out_size, void* d_ws, size_t ws_size,
                              hipStream_t stream) {
    const float* inp  = (const float*)d_in[0];
    const int*   zm   = (const int*)d_in[1];
    const float* cb   = (const float*)d_in[2];
    const float* lf32 = (const float*)d_in[3];
    const float* ls32 = (const float*)d_in[4];
    const float* gs   = (const float*)d_in[5];
    const float* gb   = (const float*)d_in[6];
    const float* sc   = (const float*)d_in[7];
    const float* bi   = (const float*)d_in[8];
    float* out = (float*)d_out;

    char* ws = (char*)d_ws;
    __bf16* act = (__bf16*)(ws);
    __bf16* wq  = (__bf16*)(ws + 67108864);
    __bf16* lf  = (__bf16*)(ws + 100663296);
    __bf16* ls  = (__bf16*)(ws + 101711872);
    __bf16* hid = (__bf16*)(ws + 103809024);

    // 1) conversions to bf16
    k_f32_to_bf16<<<(TOKENS * IN_F) / 2048, 256, 0, stream>>>(inp, act, TOKENS * IN_F);
    k_f32_to_bf16<<<(LRK * IN_F) / 2048, 256, 0, stream>>>(lf32, lf, LRK * IN_F);
    k_f32_to_bf16<<<(2 * OUT_F * LRK) / 2048, 256, 0, stream>>>(ls32, ls, 2 * OUT_F * LRK);

    // 2) dequantize weight to bf16
    k_dequant<<<(OUT_F * IN_F / 8) / 256, 256, 0, stream>>>(zm, cb, gs, gb, wq);

    // 3) low-rank hidden: hid = act @ lf^T
    k_hid_gemm<<<((TOKENS / 16) * (LRK / 16)) / 8, 256, 0, stream>>>(act, lf, hid);

    // 4) fused baseline GEMM + low-rank epilogue
    dim3 grid(OUT_F / 128, TOKENS / 128);
    k_fused_gemm<<<grid, 256, 0, stream>>>(act, wq, hid, ls, sc, bi, out);
}
